// TFAttention_59347858096454
// MI455X (gfx1250) — matrix operand order repaired and checked by reading, not compiled, not run on hardware
//
#include <hip/hip_runtime.h>

typedef __attribute__((ext_vector_type(16))) __bf16 v16bf;
typedef __attribute__((ext_vector_type(8)))  float  v8f;

union Frag {
  v16bf bf;
  uint4 u[2];
};

__device__ __forceinline__ unsigned short f32_to_bf16(float f) {
  unsigned int u = __float_as_uint(f);
  u += 0x7FFFu + ((u >> 16) & 1u);   // round-to-nearest-even
  return (unsigned short)(u >> 16);
}

__device__ __forceinline__ v8f wmma_bf16(const Frag& a, const Frag& b, v8f c) {
  return __builtin_amdgcn_wmma_f32_16x16x32_bf16(
      /*neg_a=*/false, a.bf, /*neg_b=*/false, b.bf,
      /*c_mod=*/(short)0, c, /*reuse_a=*/false, /*reuse_b=*/false);
}

// Async global -> LDS copy, 16B per lane (ASYNCcnt tracked).
// VDST VGPR carries the 32-bit LDS offset (= low 32 bits of the generic
// shared-memory address, per flat->LDS truncation in the ISA).
__device__ __forceinline__ void async_load_b128(void* lds, const void* gptr) {
  unsigned loff = (unsigned)(unsigned long long)lds;
  asm volatile("global_load_async_to_lds_b128 %0, %1, off"
               :: "v"(loff), "v"(gptr)
               : "memory");
}

__device__ __forceinline__ void wait_async0() {
  asm volatile("s_wait_asynccnt 0" ::: "memory");
}

// ---------------------------------------------------------------------------
// fp32 -> bf16 elementwise (x)
// ---------------------------------------------------------------------------
__global__ __launch_bounds__(256) void cvt_f32_bf16(
    const float* __restrict__ in, unsigned short* __restrict__ out, int n4) {
  int i = blockIdx.x * blockDim.x + threadIdx.x;
  if (i < n4) {
    float4 v = ((const float4*)in)[i];
    unsigned int lo = (unsigned int)f32_to_bf16(v.x) |
                      ((unsigned int)f32_to_bf16(v.y) << 16);
    unsigned int hi = (unsigned int)f32_to_bf16(v.z) |
                      ((unsigned int)f32_to_bf16(v.w) << 16);
    ((uint2*)out)[i] = make_uint2(lo, hi);
  }
}

// ---------------------------------------------------------------------------
// fp32 [K][N] -> bf16 [N][K] tiled transpose (weights)
// ---------------------------------------------------------------------------
__global__ __launch_bounds__(256) void transpose_f32_bf16(
    const float* __restrict__ in, unsigned short* __restrict__ out,
    int K, int N) {
  __shared__ float tile[32][33];
  const int bx = blockIdx.x, by = blockIdx.y;
  const int tx = threadIdx.x, ty = threadIdx.y;
  #pragma unroll
  for (int i = 0; i < 32; i += 8)
    tile[ty + i][tx] = in[(size_t)(by * 32 + ty + i) * N + bx * 32 + tx];
  __syncthreads();
  #pragma unroll
  for (int i = 0; i < 32; i += 8)
    out[(size_t)(bx * 32 + ty + i) * K + by * 32 + tx] =
        f32_to_bf16(tile[tx][ty + i]);
}

// ---------------------------------------------------------------------------
// bf16 GEMM: C[M,N] = A[M,K] * Bt[N,K]^T + bias
// block = 256 threads = 8 waves; block tile 128x256; wave tile 64x64 (4x4).
// Double-buffered LDS filled by GLOBAL_LOAD_ASYNC_TO_LDS_B128 (ASYNCcnt),
// next tile's copies overlap the current tile's 16 WMMAs.
// EPI 0: scatter QKV (Q scaled 1/8, V transposed); EPI 1: fp32 out
// ---------------------------------------------------------------------------
template <int EPI>
__global__ __launch_bounds__(256) void gemm_bf16_k(
    const unsigned short* __restrict__ A,
    const unsigned short* __restrict__ Bt,
    const float* __restrict__ bias,
    float* __restrict__ outF,
    unsigned short* __restrict__ outQ,
    unsigned short* __restrict__ outK,
    unsigned short* __restrict__ outV,
    int K) {
  __shared__ unsigned short As[2 * 128 * 32];   // 16 KB
  __shared__ unsigned short Bs[2 * 256 * 32];   // 32 KB

  const int tid   = threadIdx.x;
  const int lane  = tid & 31;
  const int wave  = tid >> 5;
  const int waveM = wave >> 2;   // 0..1  (64 rows each)
  const int waveN = wave & 3;    // 0..3  (64 cols each)
  const int lh    = lane >> 4;   // lane half
  const int ln    = lane & 15;

  const int m0 = blockIdx.y * 128;
  const int n0 = blockIdx.x * 256;

  v8f zero = {};
  v8f acc[4][4];
  #pragma unroll
  for (int i = 0; i < 4; ++i)
    #pragma unroll
    for (int j = 0; j < 4; ++j) acc[i][j] = zero;

  // 1536 x 16B units per K-step (A:512, B:1024); 6 per thread.
  auto issue_tile = [&](int k0, int buf) {
    unsigned short* Ab = As + buf * (128 * 32);
    unsigned short* Bb = Bs + buf * (256 * 32);
    #pragma unroll
    for (int i = 0; i < 6; ++i) {
      const int u = tid + i * 256;
      if (u < 512) {
        const int row = u >> 2, seg = (u & 3) * 8;
        async_load_b128(&Ab[row * 32 + seg],
                        A + (size_t)(m0 + row) * K + k0 + seg);
      } else {
        const int u2 = u - 512;
        const int row = u2 >> 2, seg = (u2 & 3) * 8;
        async_load_b128(&Bb[row * 32 + seg],
                        Bt + (size_t)(n0 + row) * K + k0 + seg);
      }
    }
  };

  issue_tile(0, 0);
  wait_async0();
  __syncthreads();

  const int nK = K >> 5;
  for (int it = 0; it < nK; ++it) {
    const int cur = it & 1;
    if (it + 1 < nK) issue_tile((it + 1) << 5, cur ^ 1);

    const unsigned short* Ab = As + cur * (128 * 32);
    const unsigned short* Bb = Bs + cur * (256 * 32);

    Frag af[4];
    #pragma unroll
    for (int mt = 0; mt < 4; ++mt) {
      const int row = waveM * 64 + mt * 16 + ln;
      af[mt].u[0] = *(const uint4*)&Ab[row * 32 + lh * 8];
      af[mt].u[1] = *(const uint4*)&Ab[row * 32 + 16 + lh * 8];
    }
    #pragma unroll
    for (int nt = 0; nt < 4; ++nt) {
      const int row = waveN * 64 + nt * 16 + ln;
      Frag bfr;
      bfr.u[0] = *(const uint4*)&Bb[row * 32 + lh * 8];
      bfr.u[1] = *(const uint4*)&Bb[row * 32 + 16 + lh * 8];
      #pragma unroll
      for (int mt = 0; mt < 4; ++mt)
        acc[mt][nt] = wmma_bf16(af[mt], bfr, acc[mt][nt]);
    }

    wait_async0();
    __syncthreads();
  }

  #pragma unroll
  for (int mt = 0; mt < 4; ++mt) {
    #pragma unroll
    for (int nt = 0; nt < 4; ++nt) {
      const int n = n0 + waveN * 64 + nt * 16 + ln;
      const float bn = bias[n];
      if constexpr (EPI == 0) {
        const int part = n >> 10;       // 0=q 1=k 2=v
        const int d    = n & 1023;
        const int head = d >> 6;
        const int e    = d & 63;
        #pragma unroll
        for (int vr = 0; vr < 8; ++vr) {
          const int m  = m0 + waveM * 64 + mt * 16 + vr + lh * 8;
          const int b  = m >> 11;
          const int s  = m & 2047;
          const int bh = b * 16 + head;
          const float val = acc[mt][nt][vr] + bn;
          if (part == 0)
            outQ[((size_t)bh * 2048 + s) * 64 + e] = f32_to_bf16(val * 0.125f);
          else if (part == 1)
            outK[((size_t)bh * 2048 + s) * 64 + e] = f32_to_bf16(val);
          else
            outV[((size_t)bh * 64 + e) * 2048 + s] = f32_to_bf16(val);
        }
      } else {
        #pragma unroll
        for (int vr = 0; vr < 8; ++vr) {
          const int m = m0 + waveM * 64 + mt * 16 + vr + lh * 8;
          outF[(size_t)m * 1024 + n] = acc[mt][nt][vr] + bn;
        }
      }
    }
  }
}

// ---------------------------------------------------------------------------
// Flash attention: one wave = 16 q rows, 32-key chunks, online softmax.
// Q,K: [B*H][2048][64] bf16 (Q pre-scaled 1/8); Vt: [B*H][64][2048] bf16
// O: merged-heads [B*S][1024] bf16
// ---------------------------------------------------------------------------
__global__ __launch_bounds__(128) void attn_k(
    const unsigned short* __restrict__ Q,
    const unsigned short* __restrict__ Kmat,
    const unsigned short* __restrict__ Vt,
    unsigned short* __restrict__ O) {
  __shared__ unsigned short Pl[4][16 * 32];

  const int tid  = threadIdx.x;
  const int lane = tid & 31;
  const int wave = tid >> 5;
  const int lh   = lane >> 4;
  const int ln   = lane & 15;

  const int bh = blockIdx.y;
  const int q0 = (blockIdx.x * 4 + wave) * 16;

  const size_t base = (size_t)bh * 2048 * 64;

  Frag qf[2];
  {
    const unsigned short* qp = Q + base + (size_t)(q0 + ln) * 64;
    #pragma unroll
    for (int ks = 0; ks < 2; ++ks) {
      qf[ks].u[0] = *(const uint4*)(qp + ks * 32 + lh * 8);
      qf[ks].u[1] = *(const uint4*)(qp + ks * 32 + 16 + lh * 8);
    }
  }

  v8f zero = {};
  v8f o[4] = {zero, zero, zero, zero};
  float mrow[8], lrow[8];
  #pragma unroll
  for (int r0 = 0; r0 < 8; ++r0) {
    mrow[r0] = -10000.0f;
    lrow[r0] = 0.0f;
  }

  unsigned short* pl = Pl[wave];
  const int nk = q0 + 16;   // causal: need keys [0, q0+16)

  for (int kb = 0; kb < nk; kb += 32) {
    // ---- issue all global loads for this chunk up front ----
    Frag kf[2][2];
    #pragma unroll
    for (int nt = 0; nt < 2; ++nt) {
      const unsigned short* kp = Kmat + base + (size_t)(kb + nt * 16 + ln) * 64;
      #pragma unroll
      for (int ks = 0; ks < 2; ++ks) {
        kf[nt][ks].u[0] = *(const uint4*)(kp + ks * 32 + lh * 8);
        kf[nt][ks].u[1] = *(const uint4*)(kp + ks * 32 + 16 + lh * 8);
      }
    }
    Frag vf[4];   // overlap these with QK^T WMMAs + softmax VALU
    #pragma unroll
    for (int t = 0; t < 4; ++t) {
      const unsigned short* vp = Vt + (size_t)bh * 64 * 2048 +
                                 (size_t)(t * 16 + ln) * 2048 + kb + lh * 8;
      vf[t].u[0] = *(const uint4*)vp;
      vf[t].u[1] = *(const uint4*)(vp + 16);
    }

    // ---- S = Q * K^T (16x32 via two 16x16 accumulators) ----
    v8f s0 = zero, s1 = zero;
    s0 = wmma_bf16(qf[0], kf[0][0], s0);
    s0 = wmma_bf16(qf[1], kf[0][1], s0);
    s1 = wmma_bf16(qf[0], kf[1][0], s1);
    s1 = wmma_bf16(qf[1], kf[1][1], s1);

    // ---- causal mask (reference: masked logit == -10000 exactly) ----
    if (kb + 31 > q0) {
      const int j0 = kb + ln;
      const int j1 = kb + 16 + ln;
      #pragma unroll
      for (int r0 = 0; r0 < 8; ++r0) {
        const int qi = q0 + r0 + lh * 8;
        if (j0 > qi) s0[r0] = -10000.0f;
        if (j1 > qi) s1[r0] = -10000.0f;
      }
    }

    // ---- online softmax (row reductions within 16-lane halves) ----
    float scv[8];
    #pragma unroll
    for (int r0 = 0; r0 < 8; ++r0) {
      float rm = fmaxf(s0[r0], s1[r0]);
      #pragma unroll
      for (int msk = 1; msk < 16; msk <<= 1)
        rm = fmaxf(rm, __shfl_xor(rm, msk, 32));
      const float mnew = fmaxf(mrow[r0], rm);
      const float sc = __expf(mrow[r0] - mnew);
      const float p0 = __expf(s0[r0] - mnew);
      const float p1 = __expf(s1[r0] - mnew);
      s0[r0] = p0;
      s1[r0] = p1;
      float rs = p0 + p1;
      #pragma unroll
      for (int msk = 1; msk < 16; msk <<= 1)
        rs += __shfl_xor(rs, msk, 32);
      lrow[r0] = lrow[r0] * sc + rs;
      mrow[r0] = mnew;
      scv[r0] = sc;
    }
    #pragma unroll
    for (int t = 0; t < 4; ++t)
      #pragma unroll
      for (int r0 = 0; r0 < 8; ++r0) o[t][r0] *= scv[r0];

    // ---- C-layout -> A-layout transpose of P through LDS ----
    #pragma unroll
    for (int r0 = 0; r0 < 8; ++r0) {
      const int row = r0 + lh * 8;
      pl[row * 32 + ln]      = f32_to_bf16(s0[r0]);
      pl[row * 32 + 16 + ln] = f32_to_bf16(s1[r0]);
    }
    asm volatile("s_wait_dscnt 0" ::: "memory");  // same-wave LDS RAW
    Frag pf;
    pf.u[0] = *(const uint4*)&pl[ln * 32 + lh * 8];
    pf.u[1] = *(const uint4*)&pl[ln * 32 + 16 + lh * 8];

    // ---- O += P * V ----
    #pragma unroll
    for (int t = 0; t < 4; ++t) o[t] = wmma_bf16(pf, vf[t], o[t]);
  }

  // ---- normalize + merged-heads store ----
  const int b  = bh >> 4;
  const int hh = bh & 15;
  #pragma unroll
  for (int t = 0; t < 4; ++t) {
    #pragma unroll
    for (int r0 = 0; r0 < 8; ++r0) {
      const float val = o[t][r0] / lrow[r0];
      const size_t m  = (size_t)b * 2048 + q0 + r0 + lh * 8;
      const int col   = hh * 64 + t * 16 + ln;
      O[m * 1024 + col] = f32_to_bf16(val);
    }
  }
}

// ---------------------------------------------------------------------------
extern "C" void kernel_launch(void* const* d_in, const int* in_sizes, int n_in,
                              void* d_out, int out_size, void* d_ws,
                              size_t ws_size, hipStream_t stream) {
  const float* x  = (const float*)d_in[0];
  const float* wq = (const float*)d_in[1];   // [1024][3072]
  const float* bq = (const float*)d_in[2];   // [3072]
  const float* wp = (const float*)d_in[3];   // [1024][1024]
  const float* bp = (const float*)d_in[4];   // [1024]
  float* out = (float*)d_out;

  char* ws = (char*)d_ws;
  unsigned short* Xb    = (unsigned short*)ws; ws += (size_t)8192 * 1024 * 2;
  unsigned short* Wqkv  = (unsigned short*)ws; ws += (size_t)3072 * 1024 * 2;
  unsigned short* Wproj = (unsigned short*)ws; ws += (size_t)1024 * 1024 * 2;
  unsigned short* Qb    = (unsigned short*)ws; ws += (size_t)64 * 2048 * 64 * 2;
  unsigned short* Kb    = (unsigned short*)ws; ws += (size_t)64 * 2048 * 64 * 2;
  unsigned short* Vtb   = (unsigned short*)ws; ws += (size_t)64 * 64 * 2048 * 2;
  unsigned short* Ab    = (unsigned short*)ws; ws += (size_t)8192 * 1024 * 2;

  cvt_f32_bf16<<<8192, 256, 0, stream>>>(x, Xb, 8192 * 1024 / 4);
  transpose_f32_bf16<<<dim3(96, 32), dim3(32, 8), 0, stream>>>(wq, Wqkv, 1024, 3072);
  transpose_f32_bf16<<<dim3(32, 32), dim3(32, 8), 0, stream>>>(wp, Wproj, 1024, 1024);

  // QKV projection: M=8192, N=3072, K=1024
  gemm_bf16_k<0><<<dim3(12, 64), 256, 0, stream>>>(Xb, Wqkv, bq, nullptr, Qb,
                                                   Kb, Vtb, 1024);
  // attention: 64 (b,h) pairs x 128 q-tiles (4 per block)
  attn_k<<<dim3(32, 64), 128, 0, stream>>>(Qb, Kb, Vtb, Ab);
  // output projection: M=8192, N=1024, K=1024
  gemm_bf16_k<1><<<dim3(4, 64), 256, 0, stream>>>(Ab, Wproj, bp, out, nullptr,
                                                  nullptr, nullptr, 1024);
}
